// FLAMamba_5351529251275
// MI455X (gfx1250) — compile-verified, matrix operand order checked by reading, not run on hardware
//
#include <hip/hip_runtime.h>
#include <hip/hip_bf16.h>
#include <math.h>

// ---------------------------------------------------------------------------
// Mamba block forward for MI455X (gfx1250, wave32, WMMA).
// Sizes fixed by the reference: B=1, L=2048, D_MODEL=1024, D_INNER=2048,
// D_STATE=16, D_CONV=4, DT_RANK=64.
// ---------------------------------------------------------------------------

#define L_SEQ   2048
#define DMODEL  1024
#define DINNER  2048
#define DSTATE  16
#define DTRANK  64

typedef __attribute__((ext_vector_type(16))) __bf16 v16bf;
typedef __attribute__((ext_vector_type(8)))  float  v8f;

// Async global->LDS copy path (ASYNCcnt), if the toolchain exposes it.
#if defined(__has_builtin)
#if __has_builtin(__builtin_amdgcn_global_load_async_to_lds_b128) && \
    __has_builtin(__builtin_amdgcn_s_wait_asynccnt)
#define USE_ASYNC_LDS 1
#endif
#endif
#ifndef USE_ASYNC_LDS
#define USE_ASYNC_LDS 0
#endif

#if USE_ASYNC_LDS
// Per the clang-22 diagnostic, the b128 builtin takes typed vector pointers:
// param0: int __attribute__((vector_size(16)))* (generic/global side)
// param1: LDS side (address space 3)
typedef int async_v4i __attribute__((vector_size(16)));
typedef __attribute__((address_space(3))) async_v4i* lds_v4i_ptr;
#endif

// -------------------------- helpers ----------------------------------------

static __device__ __forceinline__ unsigned short f32_to_bf16_rne(float f) {
    unsigned int u = __float_as_uint(f);
    unsigned int lsb = (u >> 16) & 1u;
    u += 0x7fffu + lsb;
    return (unsigned short)(u >> 16);
}

union FragU {
    v16bf v;
    uint4 q[2];
};

// Load one 16x32 bf16 WMMA operand fragment for this lane.
// p points at (row, k + half*8); elements 0..7 = k..k+7, 8..15 = k+16..k+23.
static __device__ __forceinline__ v16bf ldfrag(const unsigned short* p) {
    FragU f;
    f.q[0] = *(const uint4*)(p);
    f.q[1] = *(const uint4*)(p + 16);
    return f.v;
}

static __device__ __forceinline__ float softplus_f(float v) {
    return (v > 20.0f) ? v : log1pf(__expf(v));
}

static __device__ __forceinline__ void store_tile(
    float* __restrict__ C, int N, v8f c, int mbase, int nbase,
    int r, int half, const float* __restrict__ bias, int epi)
{
    const int col = nbase + r;
    if (col >= N) return;
    float b = (epi == 1) ? bias[col] : 0.0f;
    #pragma unroll
    for (int i = 0; i < 8; ++i) {
        int row = mbase + half * 8 + i;
        float v = c[i];
        if (epi == 1) v = softplus_f(v + b);
        C[(size_t)row * (size_t)N + (size_t)col] = v;
    }
}

// -------------------------- f32 -> bf16 conversion --------------------------

__global__ void mamba_cvt_bf16_kernel(const float* __restrict__ in,
                                      unsigned short* __restrict__ out, int n) {
    int i = blockIdx.x * blockDim.x + threadIdx.x;
    int stride = gridDim.x * blockDim.x;
    for (; i < n; i += stride) out[i] = f32_to_bf16_rne(in[i]);
}

// -------------------------- WMMA GEMM (NT) ----------------------------------
// C[M x N] (f32, row-major, ldc=N) = A[M x K] (bf16 rows over k)
//                                  x B[N x K] (bf16 rows over k, i.e. W row-major)
// Block: 256 threads = 8 waves; waves tiled 4(M) x 2(N); wave tile 32 x 64
// (2 A-frags x 4 B-frags -> 8 WMMAs per K-step). Block tile: 128(M) x 128(N).
// Requires M % 128 == 0, K % 32 == 0. Columns >= N are guarded.
// epi==1: C = softplus(C + bias[n]).

__global__ void __launch_bounds__(256)
mamba_wmma_gemm_bf16_kernel(const unsigned short* __restrict__ A,
                            const unsigned short* __restrict__ B,
                            float* __restrict__ C,
                            int M, int N, int K,
                            const float* __restrict__ bias, int epi)
{
    const int lane = threadIdx.x & 31;
    const int wave = threadIdx.x >> 5;      // 0..7
    const int wr   = wave >> 1;             // 0..3 (M)
    const int wc   = wave & 1;              // 0..1 (N)
    const int m0   = blockIdx.x * 128 + wr * 32;
    const int n0   = blockIdx.y * 128 + wc * 64;
    const int half = lane >> 4;             // 0/1: which K-half this lane holds
    const int r    = lane & 15;

    v8f acc[2][4];
    #pragma unroll
    for (int i = 0; i < 2; ++i)
        #pragma unroll
        for (int j = 0; j < 4; ++j)
            acc[i][j] = (v8f){};

    const unsigned short* arow[2];
    arow[0] = A + (size_t)(m0 + r)      * (size_t)K;
    arow[1] = A + (size_t)(m0 + 16 + r) * (size_t)K;

    // Columns beyond N clamp to row 0: garbage accumulates in columns that
    // are never stored (store guard below).
    const unsigned short* brow[4];
    #pragma unroll
    for (int j = 0; j < 4; ++j) {
        const int n = n0 + j * 16 + r;
        brow[j] = B + (size_t)((n < N) ? n : 0) * (size_t)K;
    }

    const int koff = half * 8;

    for (int k = 0; k < K; k += 32) {
        // Prefetch next K-slab (speculative; dropped if OOB).
        __builtin_prefetch((const void*)(arow[0] + k + 32 + koff), 0, 0);
        __builtin_prefetch((const void*)(brow[0] + k + 32 + koff), 0, 0);

        v16bf a[2], b[4];
        #pragma unroll
        for (int i = 0; i < 2; ++i) a[i] = ldfrag(arow[i] + k + koff);
        #pragma unroll
        for (int j = 0; j < 4; ++j) b[j] = ldfrag(brow[j] + k + koff);

        #pragma unroll
        for (int i = 0; i < 2; ++i)
            #pragma unroll
            for (int j = 0; j < 4; ++j)
                acc[i][j] = __builtin_amdgcn_wmma_f32_16x16x32_bf16(
                    false, a[i], false, b[j], (short)0, acc[i][j], false, false);
    }

    #pragma unroll
    for (int i = 0; i < 2; ++i)
        #pragma unroll
        for (int j = 0; j < 4; ++j)
            store_tile(C, N, acc[i][j], m0 + i * 16, n0 + j * 16, r, half, bias, epi);
}

// ---------------- depthwise causal conv1d (k=4) + bias + SiLU ----------------
// xi = xz[:, 0:2048]; xc[t][d] = silu(sum_j w[d][j]*xi[t-3+j][d] + b[d]).
// Writes f32 xc (scan input) and bf16 xc (x_proj GEMM input).

__global__ void mamba_conv_silu_kernel(const float* __restrict__ xz,
                                       const float* __restrict__ conv_w,
                                       const float* __restrict__ conv_b,
                                       float* __restrict__ xc,
                                       unsigned short* __restrict__ xc_bf)
{
    int idx = blockIdx.x * blockDim.x + threadIdx.x;  // over L*DINNER
    if (idx >= L_SEQ * DINNER) return;
    const int d = idx & (DINNER - 1);
    const int t = idx >> 11;

    const float w0 = conv_w[d * 4 + 0];
    const float w1 = conv_w[d * 4 + 1];
    const float w2 = conv_w[d * 4 + 2];
    const float w3 = conv_w[d * 4 + 3];

    float acc = conv_b[d];
    if (t >= 3) acc += w0 * xz[(size_t)(t - 3) * 4096 + d];
    if (t >= 2) acc += w1 * xz[(size_t)(t - 2) * 4096 + d];
    if (t >= 1) acc += w2 * xz[(size_t)(t - 1) * 4096 + d];
    acc += w3 * xz[(size_t)t * 4096 + d];

    const float s = acc / (1.0f + __expf(-acc));   // SiLU
    xc[idx] = s;
    xc_bf[idx] = f32_to_bf16_rne(s);
}

// ------------- split dt (first DTRANK cols of x_dbl) to bf16 -----------------

__global__ void mamba_split_dt_kernel(const float* __restrict__ x_dbl,
                                      unsigned short* __restrict__ dt_bf)
{
    int idx = blockIdx.x * blockDim.x + threadIdx.x;  // over L*DTRANK
    if (idx >= L_SEQ * DTRANK) return;
    const int t = idx >> 6;
    const int rr = idx & (DTRANK - 1);
    dt_bf[idx] = f32_to_bf16_rne(x_dbl[(size_t)t * 96 + rr]);
}

// -------------------------- selective scan ----------------------------------
// One lane per channel d; 16 states in registers; sequential over L.
// Shared B_t / C_t vectors staged through LDS in 64-timestep chunks, using the
// CDNA5 async global->LDS path (ASYNCcnt) when available.

__global__ void __launch_bounds__(256)
mamba_scan_kernel(const float* __restrict__ delta,  // L x DINNER (softplus'ed)
                  const float* __restrict__ xc,     // L x DINNER
                  const float* __restrict__ x_dbl,  // L x 96 (B @ +64, C @ +80)
                  const float* __restrict__ A_log,  // DINNER x 16
                  const float* __restrict__ Dp,     // DINNER
                  float* __restrict__ y)            // L x DINNER
{
    __shared__ float sBC[64][32];   // [tt][0..15]=B_t, [16..31]=C_t
    const int d = blockIdx.x * 256 + threadIdx.x;

    float Arow[DSTATE];
    #pragma unroll
    for (int n = 0; n < DSTATE; ++n)
        Arow[n] = -__expf(A_log[(size_t)d * DSTATE + n]);
    const float Dd = Dp[d];

    float h[DSTATE];
    #pragma unroll
    for (int n = 0; n < DSTATE; ++n) h[n] = 0.0f;

    for (int t0 = 0; t0 < L_SEQ; t0 += 64) {
        __syncthreads();
#if USE_ASYNC_LDS
        // 64 rows x 32 floats (cols 64..95) = 512 x b128 transfers; ASYNCcnt.
        {
            #pragma unroll
            for (int j = 0; j < 2; ++j) {
                const int idx = j * 256 + threadIdx.x;   // 0..511
                const int tt = idx >> 3;
                const int q  = idx & 7;
                const float* gp = x_dbl + (size_t)(t0 + tt) * 96 + 64 + q * 4;
                __builtin_amdgcn_global_load_async_to_lds_b128(
                    (async_v4i*)gp,
                    (lds_v4i_ptr)(unsigned int)(unsigned long long)(&sBC[tt][q * 4]),
                    0, 0);
            }
            __builtin_amdgcn_s_wait_asynccnt(0);
        }
#else
        for (int i = threadIdx.x; i < 64 * 32; i += 256) {
            const int tt = i >> 5;
            const int c  = i & 31;
            sBC[tt][c] = x_dbl[(size_t)(t0 + tt) * 96 + 64 + c];
        }
#endif
        __syncthreads();

        for (int tt = 0; tt < 64; ++tt) {
            const int t = t0 + tt;
            const float dl = delta[(size_t)t * DINNER + d];
            const float u  = xc[(size_t)t * DINNER + d];
            const float du = dl * u;
            float acc = 0.0f;
            #pragma unroll
            for (int n = 0; n < DSTATE; ++n) {
                const float a = __expf(dl * Arow[n]);
                h[n] = a * h[n] + du * sBC[tt][n];
                acc += h[n] * sBC[tt][16 + n];
            }
            y[(size_t)t * DINNER + d] = acc + u * Dd;
        }
    }
}

// ------------- gate: g = y * silu(z) -> bf16 (out_proj input) ----------------

__global__ void mamba_gate_kernel(const float* __restrict__ y,
                                  const float* __restrict__ xz,
                                  unsigned short* __restrict__ g_bf)
{
    int idx = blockIdx.x * blockDim.x + threadIdx.x;  // over L*DINNER
    if (idx >= L_SEQ * DINNER) return;
    const int t = idx >> 11;
    const int d = idx & (DINNER - 1);
    const float z = xz[(size_t)t * 4096 + DINNER + d];
    const float g = y[idx] * (z / (1.0f + __expf(-z)));
    g_bf[idx] = f32_to_bf16_rne(g);
}

// ---------------------------------------------------------------------------

extern "C" void kernel_launch(void* const* d_in, const int* in_sizes, int n_in,
                              void* d_out, int out_size, void* d_ws, size_t ws_size,
                              hipStream_t stream) {
    (void)in_sizes; (void)n_in; (void)out_size; (void)ws_size;

    const float* x         = (const float*)d_in[0];  // [L, DMODEL]
    const float* in_proj_w = (const float*)d_in[1];  // [2*DINNER, DMODEL]
    const float* conv_w    = (const float*)d_in[2];  // [DINNER, 1, 4]
    const float* conv_b    = (const float*)d_in[3];  // [DINNER]
    const float* x_proj_w  = (const float*)d_in[4];  // [96, DINNER]
    const float* dt_proj_w = (const float*)d_in[5];  // [DINNER, DTRANK]
    const float* dt_proj_b = (const float*)d_in[6];  // [DINNER]
    const float* A_log     = (const float*)d_in[7];  // [DINNER, 16]
    const float* Dp        = (const float*)d_in[8];  // [DINNER]
    const float* out_proj_w= (const float*)d_in[9];  // [DMODEL, DINNER]
    float* out = (float*)d_out;                      // [L, DMODEL]

    // ---- workspace layout (all offsets 256B aligned) ----
    char* ws = (char*)d_ws;
    size_t off = 0;
    auto alloc = [&](size_t bytes) -> char* {
        char* p = ws + off;
        off += (bytes + 255) & ~(size_t)255;
        return p;
    };
    float*          xz_f    = (float*)alloc((size_t)L_SEQ * 4096 * 4);          // 32 MB
    float*          xc_f    = (float*)alloc((size_t)L_SEQ * DINNER * 4);        // 16 MB
    float*          xdbl_f  = (float*)alloc((size_t)L_SEQ * 96 * 4);            // .75 MB
    float*          delta_f = (float*)alloc((size_t)L_SEQ * DINNER * 4);        // 16 MB
    float*          y_f     = (float*)alloc((size_t)L_SEQ * DINNER * 4);        // 16 MB
    unsigned short* x_bf    = (unsigned short*)alloc((size_t)L_SEQ * DMODEL * 2);
    unsigned short* inw_bf  = (unsigned short*)alloc((size_t)2 * DINNER * DMODEL * 2);
    unsigned short* xpw_bf  = (unsigned short*)alloc((size_t)96 * DINNER * 2);
    unsigned short* dtw_bf  = (unsigned short*)alloc((size_t)DINNER * DTRANK * 2);
    unsigned short* opw_bf  = (unsigned short*)alloc((size_t)DMODEL * DINNER * 2);
    unsigned short* xc_bf   = (unsigned short*)alloc((size_t)L_SEQ * DINNER * 2);
    unsigned short* dt_bf   = (unsigned short*)alloc((size_t)L_SEQ * DTRANK * 2);
    unsigned short* g_bf    = (unsigned short*)alloc((size_t)L_SEQ * DINNER * 2);

    const int TPB = 256;
    auto cdiv = [](int a, int b) { return (a + b - 1) / b; };

    // ---- 1. convert inputs/weights to bf16 ----
    {
        struct { const float* src; unsigned short* dst; int n; } cvts[] = {
            { x,          x_bf,   L_SEQ * DMODEL },
            { in_proj_w,  inw_bf, 2 * DINNER * DMODEL },
            { x_proj_w,   xpw_bf, 96 * DINNER },
            { dt_proj_w,  dtw_bf, DINNER * DTRANK },
            { out_proj_w, opw_bf, DMODEL * DINNER },
        };
        for (auto& c : cvts) {
            int blocks = cdiv(c.n, TPB); if (blocks > 4096) blocks = 4096;
            mamba_cvt_bf16_kernel<<<blocks, TPB, 0, stream>>>(c.src, c.dst, c.n);
        }
    }

    // ---- 2. in_proj: xz[L,4096] = x[L,1024] @ in_proj_w[4096,1024]^T ----
    mamba_wmma_gemm_bf16_kernel<<<dim3(L_SEQ / 128, 4096 / 128), TPB, 0, stream>>>(
        x_bf, inw_bf, xz_f, L_SEQ, 4096, DMODEL, nullptr, 0);

    // ---- 3. depthwise conv + SiLU ----
    mamba_conv_silu_kernel<<<cdiv(L_SEQ * DINNER, TPB), TPB, 0, stream>>>(
        xz_f, conv_w, conv_b, xc_f, xc_bf);

    // ---- 4. x_proj: x_dbl[L,96] = xc[L,2048] @ x_proj_w[96,2048]^T ----
    mamba_wmma_gemm_bf16_kernel<<<dim3(L_SEQ / 128, cdiv(96, 128)), TPB, 0, stream>>>(
        xc_bf, xpw_bf, xdbl_f, L_SEQ, 96, DINNER, nullptr, 0);

    // ---- 5. split dt -> bf16 ----
    mamba_split_dt_kernel<<<cdiv(L_SEQ * DTRANK, TPB), TPB, 0, stream>>>(
        xdbl_f, dt_bf);

    // ---- 6. dt_proj + fused softplus(. + bias):
    //         delta[L,2048] = softplus(dt[L,64] @ dt_proj_w[2048,64]^T + b) ----
    mamba_wmma_gemm_bf16_kernel<<<dim3(L_SEQ / 128, DINNER / 128), TPB, 0, stream>>>(
        dt_bf, dtw_bf, delta_f, L_SEQ, DINNER, DTRANK, dt_proj_b, 1);

    // ---- 7. selective scan -> y ----
    mamba_scan_kernel<<<DINNER / 256, 256, 0, stream>>>(
        delta_f, xc_f, xdbl_f, A_log, Dp, y_f);

    // ---- 8. gate: g = y * silu(z) -> bf16 ----
    mamba_gate_kernel<<<cdiv(L_SEQ * DINNER, TPB), TPB, 0, stream>>>(
        y_f, xz_f, g_bf);

    // ---- 9. out_proj: out[L,1024] = g[L,2048] @ out_proj_w[1024,2048]^T ----
    mamba_wmma_gemm_bf16_kernel<<<dim3(L_SEQ / 128, DMODEL / 128), TPB, 0, stream>>>(
        g_bf, opw_bf, out, L_SEQ, DMODEL, DINNER, nullptr, 0);
}
